// GTO_Atten_76974403879664
// MI455X (gfx1250) — matrix operand order repaired and checked by reading, not compiled, not run on hardware
//
#include <hip/hip_runtime.h>
#include <hip/hip_bf16.h>
#include <math.h>

#define B_   4
#define N_   8192
#define C_   512
#define H_   8
#define TD_  64
#define M_   256
#define BH_  32
#define SPLITS_ 4

typedef __attribute__((ext_vector_type(16))) _Float16 v16h;
typedef __attribute__((ext_vector_type(8)))  _Float16 h8;
typedef __attribute__((ext_vector_type(8)))  float    v8f;
typedef __attribute__((ext_vector_type(4)))  float    f4;

static __device__ inline v16h mk16(h8 lo, h8 hi) {
  return __builtin_shufflevector(lo, hi, 0,1,2,3,4,5,6,7,8,9,10,11,12,13,14,15);
}
static __device__ inline v8f wmma16(v16h a, v16h b, v8f c) {
  return __builtin_amdgcn_wmma_f32_16x16x32_f16(false, a, false, b, (short)0, c, false, false);
}

// Load 8 consecutive elements as f16 vector (A-tile fragments).
template <typename T> struct Ld8;
template <> struct Ld8<float> {
  static __device__ inline h8 go(const float* p) {
    f4 a = *(const f4*)p;
    f4 b = *(const f4*)(p + 4);
    h8 r;
#pragma unroll
    for (int j = 0; j < 4; ++j) { r[j] = (_Float16)a[j]; r[4 + j] = (_Float16)b[j]; }
    return r;
  }
};
template <> struct Ld8<_Float16> {
  static __device__ inline h8 go(const _Float16* p) { return *(const h8*)p; }
};

// ---------------------------------------------------------------------------
// Pack a 512x512 row-major f32 weight (k,n) into WMMA B-tile layout, f16.
// Tile (nt, ks): lane holds col n = nt*16 + (lane&15), k = ks*32 + (lane>>4)*16 + j.
// ---------------------------------------------------------------------------
__global__ void pack_w_kernel(const float* __restrict__ w0s, const float* __restrict__ w1s,
                              const float* __restrict__ w2s, _Float16* __restrict__ dst) {
  int idx = blockIdx.x * 256 + threadIdx.x;   // 3 * 512 * 512 threads
  int mat = idx >> 18;
  int rem = idx & 262143;
  const float* src = (mat == 0) ? w0s : ((mat == 1) ? w1s : w2s);
  int tile = rem >> 9;            // nt*16 + ks
  int lane = (rem >> 4) & 31;
  int j    = rem & 15;
  int n = ((tile >> 4) << 4) + (lane & 15);
  int k = ((tile & 15) << 5) + ((lane >> 4) << 4) + j;
  dst[idx] = (_Float16)src[k * C_ + n];
}

// ---------------------------------------------------------------------------
// Y[32768 x 512] = X[32768 x 512] @ W(packed) + bias.
// One wave -> 16 rows x 64 cols (4 accumulators). 8 waves cover 512 cols.
// ---------------------------------------------------------------------------
template <typename TIN, bool OUT16>
__global__ void __launch_bounds__(256) gemm512_kernel(const TIN* __restrict__ X,
                                                      const _Float16* __restrict__ Wpk,
                                                      const float* __restrict__ bias,
                                                      void* __restrict__ Y) {
  const int wave = threadIdx.x >> 5, lane = threadIdx.x & 31;
  const int m0 = blockIdx.x << 4;
  const int row = m0 + (lane & 15);
  const int kb = (lane >> 4) << 4;     // A-layout k base
  v8f acc[4] = {v8f{}, v8f{}, v8f{}, v8f{}};
#pragma unroll 4
  for (int ks = 0; ks < 16; ++ks) {
    h8 alo = Ld8<TIN>::go(X + (size_t)row * C_ + ks * 32 + kb);
    h8 ahi = Ld8<TIN>::go(X + (size_t)row * C_ + ks * 32 + 8 + kb);
    v16h A = mk16(alo, ahi);
#pragma unroll
    for (int ct = 0; ct < 4; ++ct) {
      int nt = (wave << 2) + ct;
      v16h Bt = *(const v16h*)(Wpk + (((nt << 4) + ks) << 9) + (lane << 4));
      acc[ct] = wmma16(A, Bt, acc[ct]);
    }
  }
  const int hi8 = (lane >> 4) << 3;    // C-layout row offset
#pragma unroll
  for (int ct = 0; ct < 4; ++ct) {
    int n = (wave << 6) + (ct << 4) + (lane & 15);
    float bv = bias[n];
#pragma unroll
    for (int r = 0; r < 8; ++r) {
      float v = acc[ct][r] + bv;
      size_t o = (size_t)(m0 + hi8 + r) * C_ + n;
      if (OUT16) ((_Float16*)Y)[o] = (_Float16)v;
      else       ((float*)Y)[o] = v;
    }
  }
}

// ---------------------------------------------------------------------------
// Stage 1 (split-K flash attention): per (b,h,mg,split):
//   partial softmax(Q @ kv1^T / 8) over a 2048-wide n range, partial P @ kv1.
// kv1 chunk rows stream into LDS via GLOBAL_LOAD_ASYNC_TO_LDS_B128 (ASYNCcnt).
// ---------------------------------------------------------------------------
__global__ void __launch_bounds__(256) stage1_attn_kernel(const _Float16* __restrict__ kv1,
                                                          const float* __restrict__ Q,
                                                          float* __restrict__ ptpart,
                                                          float* __restrict__ mlpart) {
  __shared__ _Float16 kv1s[32][72];    // chunk, row-major (n_local, c)
  __shared__ _Float16 kv1sT[64][40];   // chunk, transposed (c, n_local)
  __shared__ float    Sbuf[8][16][33]; // per-wave scores
  __shared__ _Float16 Pbuf[8][16][40]; // per-wave probabilities (A-layout source)
  __shared__ float    raux[8][16];     // per-row alpha broadcast

  const int wave = threadIdx.x >> 5, lane = threadIdx.x & 31;
  const int bh = blockIdx.x >> 3;
  const int mg = (blockIdx.x >> 2) & 1;
  const int sp = blockIdx.x & 3;
  const int b = bh >> 3, h = bh & 7;
  const int m0 = (mg << 7) + (wave << 4);
  const int kb = (lane >> 4) << 3;     // A-layout k base / C-layout row offset
  const int kh = (lane >> 4) << 4;     // B-layout k base
  const int lrow = lane & 15;

  // Preload Q A-tiles, folded scale 1/sqrt(64) = 0.125
  v16h Qt[2];
  {
    const float* qp = Q + ((size_t)(h * M_ + m0 + lrow)) * TD_;
#pragma unroll
    for (int ks = 0; ks < 2; ++ks) {
      f4 a0 = *(const f4*)(qp + ks * 32 + kb);
      f4 a1 = *(const f4*)(qp + ks * 32 + kb + 4);
      f4 b0 = *(const f4*)(qp + ks * 32 + 16 + kb);
      f4 b1 = *(const f4*)(qp + ks * 32 + 16 + kb + 4);
      h8 lo, hi;
#pragma unroll
      for (int j = 0; j < 4; ++j) {
        lo[j] = (_Float16)(a0[j] * 0.125f); lo[4 + j] = (_Float16)(a1[j] * 0.125f);
        hi[j] = (_Float16)(b0[j] * 0.125f); hi[4 + j] = (_Float16)(b1[j] * 0.125f);
      }
      Qt[ks] = mk16(lo, hi);
    }
  }

  v8f acc[4] = {v8f{}, v8f{}, v8f{}, v8f{}};
  float mrun = -1e30f, lrun = 0.f;
  const int tr = threadIdx.x >> 3;
  const int tc = (threadIdx.x & 7) << 3;
  const int cbeg = sp * (N_ / 32 / SPLITS_);
  const int cend = cbeg + (N_ / 32 / SPLITS_);

  for (int chunk = cbeg; chunk < cend; ++chunk) {
    const int n0 = chunk << 5;
    __syncthreads();
    // Stage kv1 chunk: async DMA global->LDS for the row-major copy,
    // regular load + scatter for the transposed copy.
    {
      const _Float16* gsrc = kv1 + ((size_t)(b * N_ + n0 + tr)) * C_ + h * TD_ + tc;
      unsigned ldsa = (unsigned)(uintptr_t)&kv1s[tr][tc];
      asm volatile("global_load_async_to_lds_b128 %0, %1, off"
                   :: "v"(ldsa), "v"((unsigned long long)(uintptr_t)gsrc)
                   : "memory");
      h8 vsrc = *(const h8*)gsrc;
#pragma unroll
      for (int j = 0; j < 8; ++j) kv1sT[tc + j][tr] = vsrc[j];
      if (chunk + 1 < cend)
        __builtin_prefetch(kv1 + ((size_t)(b * N_ + n0 + 32 + tr)) * C_ + h * TD_ + tc, 0, 1);
      asm volatile("s_wait_asynccnt 0x0" ::: "memory");
    }
    __syncthreads();

    // S = Qs @ kv1^T for this chunk (2 tiles of 16 cols)
#pragma unroll
    for (int nt = 0; nt < 2; ++nt) {
      v8f s = v8f{};
#pragma unroll
      for (int ks = 0; ks < 2; ++ks) {
        const int nn = (nt << 4) + lrow;           // B col -> kv1s row
        h8 blo = *(const h8*)&kv1s[nn][ks * 32 + kb];
        h8 bhi = *(const h8*)&kv1s[nn][ks * 32 + 16 + kb];
        s = wmma16(Qt[ks], mk16(blo, bhi), s);
      }
#pragma unroll
      for (int r = 0; r < 8; ++r) Sbuf[wave][kb + r][(nt << 4) + lrow] = s[r];
    }
    __syncthreads();

    // Online softmax: lane handles row (lane&15); halves duplicate identically
    {
      float sv[32];
#pragma unroll
      for (int c = 0; c < 32; ++c) sv[c] = Sbuf[wave][lrow][c];
      float cmax = sv[0];
#pragma unroll
      for (int c = 1; c < 32; ++c) cmax = fmaxf(cmax, sv[c]);
      float newm = fmaxf(mrun, cmax);
      float alpha = __expf(mrun - newm);
      mrun = newm;
      float psum = 0.f;
#pragma unroll
      for (int c = 0; c < 32; ++c) {
        float p = __expf(sv[c] - newm);
        psum += p;
        Pbuf[wave][lrow][c] = (_Float16)p;
      }
      lrun = lrun * alpha + psum;
      raux[wave][lrow] = alpha;
    }
    __syncthreads();

    // Rescale accumulators, then acc += P @ V
    {
      float al[8];
#pragma unroll
      for (int r = 0; r < 8; ++r) al[r] = raux[wave][kb + r];
#pragma unroll
      for (int ct = 0; ct < 4; ++ct)
#pragma unroll
        for (int r = 0; r < 8; ++r) acc[ct][r] *= al[r];

      h8 plo = *(const h8*)&Pbuf[wave][lrow][kb];
      h8 phi = *(const h8*)&Pbuf[wave][lrow][16 + kb];
      v16h PA = mk16(plo, phi);
#pragma unroll
      for (int ct = 0; ct < 4; ++ct) {
        const int cc = (ct << 4) + lrow;           // B col -> kv1sT row
        h8 blo = *(const h8*)&kv1sT[cc][kb];
        h8 bhi = *(const h8*)&kv1sT[cc][16 + kb];
        acc[ct] = wmma16(PA, mk16(blo, bhi), acc[ct]);
      }
    }
  }

  // Emit split partials: unnormalized accumulator + per-row (m, l) stats.
  if (lane < 16) {
    size_t so = (((size_t)sp * BH_ + bh) * M_ + m0 + lrow) * 2;
    mlpart[so + 0] = mrun;
    mlpart[so + 1] = lrun;
  }
#pragma unroll
  for (int r = 0; r < 8; ++r)
#pragma unroll
    for (int ct = 0; ct < 4; ++ct)
      ptpart[(((size_t)sp * BH_ + bh) * M_ + m0 + kb + r) * TD_ + (ct << 4) + lrow] = acc[ct][r];
}

// ---------------------------------------------------------------------------
// Merge the SPLITS_ flash partials into normalized pt.
// ---------------------------------------------------------------------------
__global__ void combine_pt_kernel(const float* __restrict__ ptpart,
                                  const float* __restrict__ mlpart,
                                  float* __restrict__ pt) {
  int tid = blockIdx.x * 256 + threadIdx.x;   // BH*M*TD
  int c  = tid & 63;
  int m  = (tid >> 6) & 255;
  int bh = tid >> 14;
  float mm[SPLITS_], ll[SPLITS_];
  float newm = -1e30f;
#pragma unroll
  for (int s = 0; s < SPLITS_; ++s) {
    size_t so = (((size_t)s * BH_ + bh) * M_ + m) * 2;
    mm[s] = mlpart[so + 0];
    ll[s] = mlpart[so + 1];
    newm = fmaxf(newm, mm[s]);
  }
  float lsum = 0.f, accv = 0.f;
#pragma unroll
  for (int s = 0; s < SPLITS_; ++s) {
    float w = __expf(mm[s] - newm);
    lsum += ll[s] * w;
    accv += ptpart[(((size_t)s * BH_ + bh) * M_ + m) * TD_ + c] * w;
  }
  pt[((size_t)bh * M_ + m) * TD_ + c] = accv / lsum;
}

// ---------------------------------------------------------------------------
// kv2 = pt @ qkv2_w; write k^T and v directly in packed B-tile f16 layout.
// ---------------------------------------------------------------------------
__global__ void kv2_pack_kernel(const float* __restrict__ pt, const float* __restrict__ qkv2w,
                                _Float16* __restrict__ kTpk, _Float16* __restrict__ vpk) {
  int tid = blockIdx.x * 256 + threadIdx.x;   // BH * 256 * 128
  int c2 = tid & 127;
  int m  = (tid >> 7) & 255;
  int bh = tid >> 15;
  const float* pp = pt + ((size_t)(bh * M_ + m)) * TD_;
  float s = 0.f;
#pragma unroll 8
  for (int c = 0; c < TD_; ++c) s += pp[c] * qkv2w[c * 128 + c2];
  _Float16 hv = (_Float16)s;
  if (c2 < 64) {                     // k element (m, c2) -> k^T packed
    int nt = m >> 4, ks = c2 >> 5, cc = c2 & 31;
    int hi = (cc >> 3) & 1, j = (cc & 7) | ((cc >> 4) << 3);
    int lanei = (m & 15) | (hi << 4);
    kTpk[(size_t)(((((bh << 4) + nt) << 1) + ks) << 9) + (lanei << 4) + j] = hv;
  } else {                           // v element (m, cv) -> v packed
    int cv = c2 - 64;
    int ct = cv >> 4, ks = m >> 5, mm = m & 31;
    int hi = (mm >> 3) & 1, j = (mm & 7) | ((mm >> 4) << 3);
    int lanei = (cv & 15) | (hi << 4);
    vpk[(size_t)(((((bh << 2) + ct) << 3) + ks) << 9) + (lanei << 4) + j] = hv;
  }
}

// ---------------------------------------------------------------------------
// Stage 2: per (b,h): rw = softmax(q3 @ k^T / 8) over m=256; W = rw @ v.
// One wave = 16 n rows; 8 chunks of 32 m. B tiles stream from packed k/v (L2).
// ---------------------------------------------------------------------------
__global__ void __launch_bounds__(256) stage2_attn_kernel(const _Float16* __restrict__ q3,
                                                          const _Float16* __restrict__ kTpk,
                                                          const _Float16* __restrict__ vpk,
                                                          _Float16* __restrict__ Wout) {
  __shared__ float    Sbuf[8][16][33];
  __shared__ _Float16 Pbuf[8][16][40];
  __shared__ float    raux[8][16];

  const int wave = threadIdx.x >> 5, lane = threadIdx.x & 31;
  const int bh = blockIdx.x >> 6, ng = blockIdx.x & 63;
  const int b = bh >> 3, h = bh & 7;
  const int n0 = (ng << 7) + (wave << 4);
  const int kb = (lane >> 4) << 3;
  const int lrow = lane & 15;

  v16h At[2];
  {
    const _Float16* ap = q3 + ((size_t)(b * N_ + n0 + lrow)) * C_ + h * TD_;
#pragma unroll
    for (int ks = 0; ks < 2; ++ks) {
      h8 lo = *(const h8*)(ap + ks * 32 + kb);
      h8 hi = *(const h8*)(ap + ks * 32 + 16 + kb);
      At[ks] = mk16(lo, hi);
    }
  }

  v8f acc[4] = {v8f{}, v8f{}, v8f{}, v8f{}};
  float mrun = -1e30f, lrun = 0.f;

  for (int chunk = 0; chunk < M_ / 32; ++chunk) {
#pragma unroll
    for (int nt = 0; nt < 2; ++nt) {
      v8f s = v8f{};
#pragma unroll
      for (int ks = 0; ks < 2; ++ks) {
        int ntg = (chunk << 1) + nt;
        v16h Bt = *(const v16h*)(kTpk + (size_t)(((((bh << 4) + ntg) << 1) + ks) << 9) + (lane << 4));
        s = wmma16(At[ks], Bt, s);
      }
#pragma unroll
      for (int r = 0; r < 8; ++r) Sbuf[wave][kb + r][(nt << 4) + lrow] = s[r] * 0.125f;
    }
    __syncthreads();

    {
      float sv[32];
#pragma unroll
      for (int c = 0; c < 32; ++c) sv[c] = Sbuf[wave][lrow][c];
      float cmax = sv[0];
#pragma unroll
      for (int c = 1; c < 32; ++c) cmax = fmaxf(cmax, sv[c]);
      float newm = fmaxf(mrun, cmax);
      float alpha = __expf(mrun - newm);
      mrun = newm;
      float psum = 0.f;
#pragma unroll
      for (int c = 0; c < 32; ++c) {
        float p = __expf(sv[c] - newm);
        psum += p;
        Pbuf[wave][lrow][c] = (_Float16)p;
      }
      lrun = lrun * alpha + psum;
      raux[wave][lrow] = alpha;
    }
    __syncthreads();

    {
      float al[8];
#pragma unroll
      for (int r = 0; r < 8; ++r) al[r] = raux[wave][kb + r];
#pragma unroll
      for (int ct = 0; ct < 4; ++ct)
#pragma unroll
        for (int r = 0; r < 8; ++r) acc[ct][r] *= al[r];

      h8 plo = *(const h8*)&Pbuf[wave][lrow][kb];
      h8 phi = *(const h8*)&Pbuf[wave][lrow][16 + kb];
      v16h PA = mk16(plo, phi);
#pragma unroll
      for (int ct = 0; ct < 4; ++ct) {
        v16h Bt = *(const v16h*)(vpk + (size_t)(((((bh << 2) + ct) << 3) + chunk) << 9) + (lane << 4));
        acc[ct] = wmma16(PA, Bt, acc[ct]);
      }
    }
    __syncthreads();
  }

  raux[wave][lrow] = lrun;
  __syncthreads();
#pragma unroll
  for (int r = 0; r < 8; ++r) {
    float inv = 1.0f / raux[wave][kb + r];
#pragma unroll
    for (int ct = 0; ct < 4; ++ct)
      Wout[((size_t)(b * N_ + n0 + kb + r)) * C_ + h * TD_ + (ct << 4) + lrow] =
          (_Float16)(acc[ct][r] * inv);
  }
}

// ---------------------------------------------------------------------------
extern "C" void kernel_launch(void* const* d_in, const int* in_sizes, int n_in,
                              void* d_out, int out_size, void* d_ws, size_t ws_size,
                              hipStream_t stream) {
  const float* W0     = (const float*)d_in[0];
  const float* Q      = (const float*)d_in[1];
  const float* kv1_w  = (const float*)d_in[2];
  const float* kv1_b  = (const float*)d_in[3];
  const float* qkv2_w = (const float*)d_in[4];
  const float* q3_w   = (const float*)d_in[5];
  const float* q3_b   = (const float*)d_in[6];
  const float* proj_w = (const float*)d_in[7];
  const float* proj_b = (const float*)d_in[8];

  char* ws = (char*)d_ws;
  _Float16* kv1buf = (_Float16*)ws;                                  // 32 MB (reused as Wout)
  _Float16* q3buf  = (_Float16*)(ws + 33554432);                     // 32 MB
  _Float16* wpk    = (_Float16*)(ws + 67108864);                     // 3 * 512KB packed weights
  _Float16* kTpk   = (_Float16*)(ws + 68681728);                     // 1 MB
  _Float16* vpk    = (_Float16*)(ws + 69730304);                     // 1 MB
  float*    ptbuf  = (float*)(ws + 70778880);                        // 2 MB
  float*    ptpart = (float*)(ws + 72876032);                        // 8 MB
  float*    mlpart = (float*)(ws + 81264640);                        // 256 KB

  pack_w_kernel<<<3072, 256, 0, stream>>>(kv1_w, q3_w, proj_w, wpk);
  gemm512_kernel<float, true><<<2048, 256, 0, stream>>>(W0, wpk,          kv1_b, kv1buf);
  gemm512_kernel<float, true><<<2048, 256, 0, stream>>>(W0, wpk + 262144, q3_b,  q3buf);
  stage1_attn_kernel<<<BH_ * 2 * SPLITS_, 256, 0, stream>>>(kv1buf, Q, ptpart, mlpart);
  combine_pt_kernel<<<2048, 256, 0, stream>>>(ptpart, mlpart, ptbuf);
  kv2_pack_kernel<<<4096, 256, 0, stream>>>(ptbuf, qkv2_w, kTpk, vpk);
  stage2_attn_kernel<<<2048, 256, 0, stream>>>(q3buf, kTpk, vpk, kv1buf /* Wout */);
  gemm512_kernel<_Float16, false><<<2048, 256, 0, stream>>>(kv1buf, wpk + 524288, proj_b,
                                                            (float*)d_out);
}
